// ScaledDotProductAttention_65257733095489
// MI455X (gfx1250) — matrix operand order repaired and checked by reading, not compiled, not run on hardware
//
#include <hip/hip_runtime.h>
#include <math.h>

// CDNA5 / gfx1250 wave32 WMMA attention forward.
// Shapes: B=2, H=16, S=2048, d_k=64, fp32 in/out (bf16 internally for WMMA).

typedef __attribute__((ext_vector_type(16))) __bf16 v16bf;
typedef __attribute__((ext_vector_type(8)))  float  v8f;
typedef __attribute__((ext_vector_type(4)))  float  v4f;   // native vec for nontemporal builtin

#define S_LEN   2048
#define D_KK    64
#define SSTRIDE 2052          // 2048 + 4 floats pad (rows 16B aligned)
#define NWAVES  4
#define BLOCK   (NWAVES * 32)
#define BH      32            // B*H
#define QTILES  (S_LEN / 16)  // 128
#define CTX_ELEMS ((size_t)BH * S_LEN * D_KK)
#define NCHUNK  (S_LEN / 32)  // 64 V chunks of 32 rows
#define VCHUNK_FLOATS (32 * D_KK)      // 2048 floats = 8KB
#define VCHUNK_BYTES  (VCHUNK_FLOATS * 4)

// A-matrix (16x32 bf16) fragment, contiguous k order (same as the B frags):
// lane L, row m = L%16, hl = L>>4.
//   elems 0..15 -> k = kbase + hl*16 + i
__device__ __forceinline__ v16bf make_a_frag(const float* __restrict__ row_base,
                                             int kbase, int hl, float scale) {
  v16bf a;
  const int k0 = kbase + hl * 16;
  const int k1 = kbase + hl * 16 + 8;
#pragma unroll
  for (int i = 0; i < 8; ++i) a[i]     = (__bf16)(row_base[k0 + i] * scale);
#pragma unroll
  for (int i = 0; i < 8; ++i) a[8 + i] = (__bf16)(row_base[k1 + i] * scale);
  return a;
}

// B-matrix (32x16 bf16) fragment, lane L: col n = L&15, khalf = L>>4,
// elem i -> B[khalf*16 + i][n]. Source contiguous in k.
__device__ __forceinline__ v16bf make_b_frag_contig(const float* __restrict__ p) {
  v16bf b;
#pragma unroll
  for (int i = 0; i < 16; ++i) b[i] = (__bf16)p[i];
  return b;
}

// Same fragment but source strided in k (V chunk staged row-major in LDS).
__device__ __forceinline__ v16bf make_b_frag_stride(const float* __restrict__ p, int stride) {
  v16bf b;
#pragma unroll
  for (int i = 0; i < 16; ++i) b[i] = (__bf16)p[i * stride];
  return b;
}

// CDNA5 async DMA: GLOBAL_LOAD_ASYNC_TO_LDS_B128 (ASYNCcnt-tracked, no VGPR data).
__device__ __forceinline__ void async_stage_b128(unsigned lds_byte_off, const float* gsrc) {
  asm volatile("global_load_async_to_lds_b128 %0, %1, off"
               :: "v"(lds_byte_off), "v"(gsrc) : "memory");
}

__device__ __forceinline__ void wait_asynccnt_le4() {
#if __has_builtin(__builtin_amdgcn_s_wait_asynccnt)
  __builtin_amdgcn_s_wait_asynccnt(4);
#else
  asm volatile("s_wait_asynccnt 4" ::: "memory");
#endif
}

__global__ void __launch_bounds__(BLOCK)
attn_fwd_wmma_kernel(const float* __restrict__ Q, const float* __restrict__ K,
                     const float* __restrict__ V, float* __restrict__ ctx_out,
                     float* __restrict__ attn_out) {
  extern __shared__ float smem[];
  float* sc   = smem;                        // [16][SSTRIDE] scores/probs (~128KB)
  float* red  = smem + 16 * SSTRIDE;         // [16*8] reduction scratch
  float* vbuf = smem + 16 * SSTRIDE + 128;   // [3][32*64] async V staging ring (24KB)

  const int tid  = threadIdx.x;
  const int wave = tid >> 5;
  const int lane = tid & 31;
  const int n    = lane & 15;  // WMMA N-col / A-row (m == n for A frags)
  const int hl   = lane >> 4;

  const int  blk    = blockIdx.x;     // bh * QTILES + qtile
  const int  bh     = blk >> 7;
  const int  qtile  = blk & (QTILES - 1);
  const long q_row0 = (long)bh * S_LEN + (long)qtile * 16;

  const float* Qbase = Q + q_row0 * D_KK;
  const float* Kbase = K + (long)bh * S_LEN * D_KK;
  const float* Vbase = V + (long)bh * S_LEN * D_KK;

  // LDS byte offset of the staging ring (flat LDS aperture: offset = low 32 bits).
  const unsigned vbuf_lds = (unsigned)(unsigned long long)vbuf;

  // ---------------- Phase 1: scores = (Q * 1/8) x K^T -> LDS ----------------
  const float scale = 0.125f;  // 1/sqrt(64)
  const v16bf aq0 = make_a_frag(Qbase + n * D_KK, 0,  hl, scale);
  const v16bf aq1 = make_a_frag(Qbase + n * D_KK, 32, hl, scale);

  for (int t = wave * 32; t < wave * 32 + 32; ++t) {
    const float* Krow = Kbase + (long)(t * 16 + n) * D_KK + hl * 16;
    const v16bf b0 = make_b_frag_contig(Krow);       // d = 0..31 chunk
    const v16bf b1 = make_b_frag_contig(Krow + 32);  // d = 32..63 chunk
    v8f c = {};
    c = __builtin_amdgcn_wmma_f32_16x16x32_bf16(false, aq0, false, b0, (short)0, c, false, false);
    c = __builtin_amdgcn_wmma_f32_16x16x32_bf16(false, aq1, false, b1, (short)0, c, false, false);
    // C/D layout: VGPR r holds row (hl*8 + r), col n.
    float* out = sc + (hl * 8) * SSTRIDE + t * 16 + n;
#pragma unroll
    for (int r = 0; r < 8; ++r) out[r * SSTRIDE] = c[r];
  }
  __syncthreads();

  // Kick off async staging of V chunk 0 (slot 0) so the DMA overlaps softmax.
#pragma unroll
  for (int it = 0; it < 4; ++it) {
    const int q = tid + BLOCK * it;  // 512 b128 quads per 8KB chunk
    async_stage_b128(vbuf_lds + (unsigned)q * 16u, Vbase + (size_t)q * 4);
  }

  // ---------------- Phase 2: softmax over the 2048-wide rows ----------------
  const int row = tid >> 3;  // 16 rows, 8 threads each
  const int sub = tid & 7;
  float* srow = sc + row * SSTRIDE;

  float mx = -3.402823e38f;
  for (int c = sub; c < S_LEN; c += 8) mx = fmaxf(mx, srow[c]);
  red[row * 8 + sub] = mx;
  __syncthreads();
#pragma unroll
  for (int j = 0; j < 8; ++j) mx = fmaxf(mx, red[row * 8 + j]);

  float sum = 0.f;
  for (int c = sub; c < S_LEN; c += 8) {
    const float p = __expf(srow[c] - mx);
    srow[c] = p;
    sum += p;
  }
  __syncthreads();  // everyone done reading red as maxes
  red[row * 8 + sub] = sum;
  __syncthreads();
  float tot = 0.f;
#pragma unroll
  for (int j = 0; j < 8; ++j) tot += red[row * 8 + j];
  const float inv = 1.0f / tot;
  for (int c = sub; c < S_LEN; c += 8) srow[c] *= inv;
  __syncthreads();

  // Non-temporal v4f store of the attn block: 512MB write-once stream must
  // not evict the L2-resident K/V (32MB inside the 192MB L2).
  float* attn_base = attn_out + q_row0 * (long)S_LEN;
  for (int r = 0; r < 16; ++r) {
    const v4f* src4 = (const v4f*)(sc + r * SSTRIDE);
    v4f*       dst4 = (v4f*)(attn_base + (long)r * S_LEN);
    for (int j = tid; j < S_LEN / 4; j += BLOCK)
      __builtin_nontemporal_store(src4[j], dst4 + j);
  }

  // ------- Phase 3: context = P x V, V chunks via async-DMA LDS ring --------
  // Wave w owns d-cols [w*16, w*16+16). 3-slot ring + 1 barrier/chunk is
  // race-free: writer of slot (i+1)%3 runs post-barrier_{i-1}; its previous reader
  // (compute_{i-2}) finished pre-barrier_{i-1}.
  const int dcol = wave * 16 + n;
  v8f acc = {};
  for (int i = 0; i < NCHUNK; ++i) {
    // Stage next chunk (wraps at the end; the extra 8KB re-fetch is harmless
    // and keeps ASYNCcnt bookkeeping uniform: always 4 in flight after issue).
    const int nxt  = (i + 1) & (NCHUNK - 1);
    const int slot = (i + 1) % 3;
    const float* gch = Vbase + (size_t)nxt * VCHUNK_FLOATS;
    const unsigned lbase = vbuf_lds + (unsigned)slot * VCHUNK_BYTES;
#pragma unroll
    for (int it = 0; it < 4; ++it) {
      const int q = tid + BLOCK * it;
      async_stage_b128(lbase + (unsigned)q * 16u, gch + (size_t)q * 4);
    }
    wait_asynccnt_le4();   // chunk i fully landed in LDS (in-order ASYNCcnt)
    __syncthreads();       // ...from every wave

    const float* vb = vbuf + (i % 3) * VCHUNK_FLOATS;  // [32][64]
    const v16bf ap = make_a_frag(sc + n * SSTRIDE, i * 32, hl, 1.0f);
    const v16bf bv = make_b_frag_stride(vb + (hl * 16) * D_KK + dcol, D_KK);
    acc = __builtin_amdgcn_wmma_f32_16x16x32_bf16(false, ap, false, bv, (short)0, acc, false, false);
  }

  float* ctx = ctx_out + q_row0 * D_KK;
#pragma unroll
  for (int r = 0; r < 8; ++r)
    ctx[(long)(hl * 8 + r) * D_KK + dcol] = acc[r];
}

extern "C" void kernel_launch(void* const* d_in, const int* in_sizes, int n_in,
                              void* d_out, int out_size, void* d_ws, size_t ws_size,
                              hipStream_t stream) {
  (void)in_sizes; (void)n_in; (void)out_size; (void)d_ws; (void)ws_size;

  const float* Q = (const float*)d_in[0];
  const float* K = (const float*)d_in[1];
  const float* V = (const float*)d_in[2];
  float* ctx  = (float*)d_out;              // context first in return order
  float* attn = (float*)d_out + CTX_ELEMS;  // then the attention matrix

  // sc (16*SSTRIDE) + red (128) + 3-slot V ring (3*2048) floats ~= 156KB.
  const size_t shmem = (size_t)(16 * SSTRIDE + 128 + 3 * VCHUNK_FLOATS) * sizeof(float);
  (void)hipFuncSetAttribute((const void*)attn_fwd_wmma_kernel,
                            hipFuncAttributeMaxDynamicSharedMemorySize, (int)shmem);

  dim3 grid(BH * QTILES);  // 4096 workgroups
  dim3 block(BLOCK);       // 4 wave32 waves
  attn_fwd_wmma_kernel<<<grid, block, shmem, stream>>>(Q, K, V, ctx, attn);
}
